// NNUE_32014686224968
// MI455X (gfx1250) — compile-verified, matrix operand order checked by reading, not run on hardware
//
#include <hip/hip_runtime.h>

typedef __attribute__((ext_vector_type(16))) _Float16 v16h;
typedef __attribute__((ext_vector_type(8)))  float    v8f;

#define NW      4        // waves per block
#define NFRAGS  22       // packed weight fragments: Ww' 8, Wb' 8, W0 4, W1 2
// per-wave stage (halves): layer-0 A frags [0,1024), layer-1 A frags [1024,1536)
// every slot is rewritten each tile -> no cross-tile invariants.
#define X_OFF   1024
#define STAGE   1536

// CDNA5 wave32 f16 A/B fragment mapping:  K = (v%4)*2 + h + g*8 + (v/4)*16
__device__ __forceinline__ int kOf(int v, int h, int g) {
  return ((v & 3) << 1) + h + (g << 3) + ((v >> 2) << 4);
}
// flat half-offset of element (m,k) minus the m*16 term
__device__ __forceinline__ int fragOff(int k) {
  return ((k >> 5) << 9) | ((k & 8) << 5) | ((k & 16) >> 1) | (k & 7);
}

__device__ __forceinline__ v8f wmma_f16(v16h a, v16h b, v8f c) {
  return __builtin_amdgcn_wmma_f32_16x16x32_f16(false, a, false, b, (short)0, c,
                                                false, false);
}
__device__ __forceinline__ v16h loadFrag(const _Float16* p, int lane) {
  return *(const v16h*)(p + lane * 16);   // two ds_load_b128
}
__device__ __forceinline__ float4 ld4u(const float* p) {  // 4B-aligned vector load
  float4 v; __builtin_memcpy(&v, p, sizeof(float4)); return v;
}
__device__ __forceinline__ v16h cvt16(const float* f) {   // 8x v_cvt_pk_f16_f32
  v16h a;
#pragma unroll
  for (int e = 0; e < 16; ++e) a[e] = (_Float16)f[e];
  return a;
}

__global__ __launch_bounds__(NW * 32) void nnue_kernel(
    const float* __restrict__ pov,   const float* __restrict__ white,
    const float* __restrict__ black, const float* __restrict__ Ww,
    const float* __restrict__ bw,    const float* __restrict__ Wb,
    const float* __restrict__ bb,    const float* __restrict__ W0,
    const float* __restrict__ b0,    const float* __restrict__ W1,
    const float* __restrict__ b1,    const float* __restrict__ W2,
    const float* __restrict__ b2,    float* __restrict__ out, int nrows) {
  __shared__ __align__(32) _Float16 wFragLds[NFRAGS * 512];   // 22.0 KB
  __shared__ float biasLds[32 + 32 + 32 + 32 + 64 + 1];
  __shared__ __align__(32) _Float16 stageLds[NW][STAGE];       // 12.0 KB
  __shared__ float povLds[NW][16];

  const int tid    = threadIdx.x;
  const int lane   = tid & 31;
  const int wv     = tid >> 5;
  const int g      = lane >> 4;
  const int laneLo = lane & 15;

  // ---- one-time: pack weights into WMMA B-fragment layout (f16).
  // Feature A uses a padded K axis: white at k[0,49), black at k[64,113),
  // zeros elsewhere. Ww' maps columns accordingly; Wb' additionally applies
  // the white<->black swap so both feature GEMMs share one A operand.
  for (int idx = tid; idx < NFRAGS * 512; idx += blockDim.x) {
    int f = idx >> 9;
    int L = (idx >> 4) & 31;
    int e = idx & 15;
    int k = kOf(e >> 1, e & 1, L >> 4);
    float val = 0.f;
    if (f < 8) {                              // Ww', K=128 padded
      int tn = f & 1, kc = f >> 1, n = tn * 16 + (L & 15);
      int kg = kc * 32 + k;
      if (kg < 49)                     val = Ww[n * 98 + kg];
      else if (kg >= 64 && kg < 113)   val = Ww[n * 98 + (kg - 15)];
    } else if (f < 16) {                      // Wb' (swapped), K=128 padded
      int tn = (f - 8) & 1, kc = (f - 8) >> 1, n = tn * 16 + (L & 15);
      int kg = kc * 32 + k;
      if (kg < 49)                     val = Wb[n * 98 + (kg + 49)];
      else if (kg >= 64 && kg < 113)   val = Wb[n * 98 + (kg - 64)];
    } else if (f < 20) {                      // W0, K=64
      int tn = (f - 16) & 1, kc = (f - 16) >> 1, n = tn * 16 + (L & 15);
      int kg = kc * 32 + k;
      if (kg < 64) val = W0[n * 64 + kg];
    } else {                                  // W1, K=32
      int tn = f - 20, n = tn * 16 + (L & 15);
      if (k < 32) val = W1[n * 32 + k];
    }
    wFragLds[idx] = (_Float16)val;
  }
  for (int i = tid; i < 193; i += blockDim.x) {
    float v;
    if (i < 32)       v = bw[i];
    else if (i < 64)  v = bb[i - 32];
    else if (i < 96)  v = b0[i - 64];
    else if (i < 128) v = b1[i - 96];
    else if (i < 192) v = W2[i - 128];
    else              v = b2[0];
    biasLds[i] = v;
  }
  __syncthreads();

  const float* bwS = biasLds;       const float* bbS = biasLds + 32;
  const float* b0S = biasLds + 64;  const float* b1S = biasLds + 96;
  const float* W2S = biasLds + 128; const float  b2v = biasLds[192];

  _Float16* stage = stageLds[wv];
  float*    povS  = povLds[wv];

  // loop-invariant restage bases (blend -> layer0 A, x -> layer1 A)
  const int bo0 = fragOff(laneLo)      + (g << 7);
  const int bo1 = fragOff(laneLo + 16) + (g << 7);
  const int bo2 = fragOff(laneLo + 32) + (g << 7);
  const int bo3 = fragOff(laneLo + 48) + (g << 7);
  const int xo0 = X_OFF + bo0;
  const int xo1 = X_OFF + bo1;

  const int nTiles = nrows >> 4;
  const int wave0  = blockIdx.x * NW + wv;
  const int nWaves = gridDim.x * NW;

  for (int t = wave0; t < nTiles; t += nWaves) {
    const int row0 = t << 4;

    // prefetch next tile's feature rows (global_prefetch_b8)
    if (t + nWaves < nTiles) {
      const float* wnx = white + (size_t)(t + nWaves) * 784 + lane * 32;
      const float* bnx = black + (size_t)(t + nWaves) * 784 + lane * 32;
      __builtin_prefetch(wnx, 0, 1);
      __builtin_prefetch(bnx, 0, 1);
    }

    // ---- build feature A frags in registers: per-lane row loads at
    //      compile-time offsets (lane-half shift folded into base pointer) ----
    const float* wrow = white + (size_t)(row0 + laneLo) * 49;
    const float* brow = black + (size_t)(row0 + laneLo) * 49;
    const float* wrg  = wrow + 8 * g;
    const float* brg  = brow + 8 * g;
    float4 w0_ = ld4u(wrg + 0),  w1_ = ld4u(wrg + 4);
    float4 w2_ = ld4u(wrg + 16), w3_ = ld4u(wrg + 20);
    float4 w4_ = ld4u(wrg + 32), w5_ = ld4u(wrg + 36);
    float4 k0_ = ld4u(brg + 0),  k1_ = ld4u(brg + 4);
    float4 k2_ = ld4u(brg + 16), k3_ = ld4u(brg + 20);
    float4 k4_ = ld4u(brg + 32), k5_ = ld4u(brg + 36);
    float w48t = wrow[48], b48t = brow[48];
    float w48 = g ? 0.f : w48t;      // col 48 belongs to lane-half 0 only
    float b48 = g ? 0.f : b48t;
    povS[laneLo] = pov[row0 + laneLo];

    v16h a0, a1, a2, a3;
    {
      float ta[16] = {w0_.x, w0_.y, w0_.z, w0_.w, w1_.x, w1_.y, w1_.z, w1_.w,
                      w2_.x, w2_.y, w2_.z, w2_.w, w3_.x, w3_.y, w3_.z, w3_.w};
      a0 = cvt16(ta);
      float tb[16] = {w4_.x, w4_.y, w4_.z, w4_.w, w5_.x, w5_.y, w5_.z, w5_.w,
                      w48, 0.f, 0.f, 0.f, 0.f, 0.f, 0.f, 0.f};
      a1 = cvt16(tb);
      float tc[16] = {k0_.x, k0_.y, k0_.z, k0_.w, k1_.x, k1_.y, k1_.z, k1_.w,
                      k2_.x, k2_.y, k2_.z, k2_.w, k3_.x, k3_.y, k3_.z, k3_.w};
      a2 = cvt16(tc);
      float td[16] = {k4_.x, k4_.y, k4_.z, k4_.w, k5_.x, k5_.y, k5_.z, k5_.w,
                      b48, 0.f, 0.f, 0.f, 0.f, 0.f, 0.f, 0.f};
      a3 = cvt16(td);
    }

    // ---- feature transform: one A operand, Ww' and Wb' B operands ----
    v8f cw0 = {}, cw1 = {}, cb0 = {}, cb1 = {};
    {
      v16h ach[4] = {a0, a1, a2, a3};
#pragma unroll
      for (int kc = 0; kc < 4; ++kc) {
        cw0 = wmma_f16(ach[kc], loadFrag(wFragLds + (kc * 2 + 0) * 512, lane), cw0);
        cw1 = wmma_f16(ach[kc], loadFrag(wFragLds + (kc * 2 + 1) * 512, lane), cw1);
        cb0 = wmma_f16(ach[kc], loadFrag(wFragLds + (8 + kc * 2 + 0) * 512, lane), cb0);
        cb1 = wmma_f16(ach[kc], loadFrag(wFragLds + (8 + kc * 2 + 1) * 512, lane), cb1);
      }
    }

    // ---- bias + pov blend + relu -> layer0 A frags in stage[0,1024) ----
    const float bwa = bwS[laneLo], bwb = bwS[16 + laneLo];
    const float bba = bbS[laneLo], bbb = bbS[16 + laneLo];
#pragma unroll
    for (int r = 0; r < 8; ++r) {
      float p = povS[r + 8 * g], q = 1.f - p;
      float w0v = cw0[r] + bwa, w1v = cw1[r] + bwb;
      float v0  = cb0[r] + bba, v1  = cb1[r] + bbb;
      int   ro  = r << 4;
      stage[bo0 + ro] = (_Float16)fmaxf(0.f, p * w0v + q * v0);
      stage[bo1 + ro] = (_Float16)fmaxf(0.f, p * w1v + q * v1);
      stage[bo2 + ro] = (_Float16)fmaxf(0.f, p * v0 + q * w0v);
      stage[bo3 + ro] = (_Float16)fmaxf(0.f, p * v1 + q * w1v);
    }

    // ---- layer 0: [16x64] x [64x32], relu ----
    v8f cx0 = {}, cx1 = {};
#pragma unroll
    for (int kc = 0; kc < 2; ++kc) {
      v16h a = loadFrag(stage + kc * 512, lane);
      cx0 = wmma_f16(a, loadFrag(wFragLds + (16 + kc * 2 + 0) * 512, lane), cx0);
      cx1 = wmma_f16(a, loadFrag(wFragLds + (16 + kc * 2 + 1) * 512, lane), cx1);
    }
    const float c00 = b0S[laneLo], c01 = b0S[16 + laneLo];
#pragma unroll
    for (int r = 0; r < 8; ++r) {
      cx0[r] = fmaxf(0.f, cx0[r] + c00);
      cx1[r] = fmaxf(0.f, cx1[r] + c01);
      int ro = r << 4;
      stage[xo0 + ro] = (_Float16)cx0[r];   // layer1 A frags in stage[1024,1536)
      stage[xo1 + ro] = (_Float16)cx1[r];
    }

    // ---- layer 1: [16x32] x [32x32], relu ----
    v8f cy0 = {}, cy1 = {};
    {
      v16h a = loadFrag(stage + X_OFF, lane);
      cy0 = wmma_f16(a, loadFrag(wFragLds + 20 * 512, lane), cy0);
      cy1 = wmma_f16(a, loadFrag(wFragLds + 21 * 512, lane), cy1);
    }
    const float c10 = b1S[laneLo], c11 = b1S[16 + laneLo];

    // ---- head: cat(x,y) . W2 + b2, butterfly over the 16 N-lanes ----
    const float w2a = W2S[laneLo],      w2b = W2S[16 + laneLo];
    const float w2c = W2S[32 + laneLo], w2d = W2S[48 + laneLo];
    float s[8];
#pragma unroll
    for (int r = 0; r < 8; ++r) {
      float y0 = fmaxf(0.f, cy0[r] + c10);
      float y1 = fmaxf(0.f, cy1[r] + c11);
      s[r] = cx0[r] * w2a + cx1[r] * w2b + y0 * w2c + y1 * w2d;
    }
#pragma unroll
    for (int r = 0; r < 8; ++r) {
      float v = s[r];
      v += __shfl_xor(v, 1, 32);
      v += __shfl_xor(v, 2, 32);
      v += __shfl_xor(v, 4, 32);
      v += __shfl_xor(v, 8, 32);
      s[r] = v;
    }
    if (laneLo == 0) {
#pragma unroll
      for (int r = 0; r < 8; ++r) out[row0 + 8 * g + r] = s[r] + b2v;
    }
  }
}

extern "C" void kernel_launch(void* const* d_in, const int* in_sizes, int n_in,
                              void* d_out, int out_size, void* d_ws, size_t ws_size,
                              hipStream_t stream) {
  const float* pov   = (const float*)d_in[0];
  const float* white = (const float*)d_in[1];
  const float* black = (const float*)d_in[2];
  const float* Ww    = (const float*)d_in[3];
  const float* bw    = (const float*)d_in[4];
  const float* Wb    = (const float*)d_in[5];
  const float* bb    = (const float*)d_in[6];
  const float* W0    = (const float*)d_in[7];
  const float* b0    = (const float*)d_in[8];
  const float* W1    = (const float*)d_in[9];
  const float* b1    = (const float*)d_in[10];
  const float* W2    = (const float*)d_in[11];
  const float* b2    = (const float*)d_in[12];
  float* out = (float*)d_out;

  int nrows = in_sizes[0];          // pov is [B,1]
  int tiles = nrows >> 4;
  int blocks = 4096;                // 4 waves/block -> 16384 waves, ~4 tiles/wave
  int maxBlocks = (tiles + NW - 1) / NW;
  if (blocks > maxBlocks) blocks = maxBlocks;
  if (blocks < 1) blocks = 1;

  nnue_kernel<<<blocks, NW * 32, 0, stream>>>(pov, white, black, Ww, bw, Wb, bb,
                                              W0, b0, W1, b1, W2, b2, out, nrows);
}
